// NNLoss_65584150610029
// MI455X (gfx1250) — matrix operand order repaired and checked by reading, not compiled, not run on hardware
//
#include <hip/hip_runtime.h>

typedef _Float16 half_t;
typedef half_t v16h __attribute__((ext_vector_type(16)));
typedef half_t v8h  __attribute__((ext_vector_type(8)));
typedef float  v8f  __attribute__((ext_vector_type(8)));

#define HW     128
#define PATCH  5
#define OW     124            // 128 - 5 + 1
#define NPATCH (OW * OW)      // 15376 = 961 * 16
#define NPAD   15392          // 962 * 16 : X rows padded to a multiple of 32
#define KF     75             // 3 * 5 * 5
#define KP     96             // padded K: 3 WMMA steps of 32
#define NT     (NPATCH / 16)  // 961 column tiles (Y side, exact)
#define NBLK   (NPAD / 32)    // 481 row blocks of 32 rows

// ---------------------------------------------------------------------------
// Kernel 1: unfold both images into f16 patch matrices (stride KP, zero
// padded K) + f32 squared row norms on the f16-rounded values. X side is
// additionally zero-padded to NPAD rows so row blocks of 32 divide evenly.
// ---------------------------------------------------------------------------
__global__ void unfold_kernel(const float* __restrict__ crop,
                              const float* __restrict__ orig,
                              half_t* __restrict__ Xh, half_t* __restrict__ Yh,
                              float* __restrict__ X2, float* __restrict__ Y2) {
    int p = blockIdx.x * blockDim.x + threadIdx.x;
    const bool xside = (blockIdx.y == 0);
    const int limit = xside ? NPAD : NPATCH;
    if (p >= limit) return;

    const float* src = xside ? crop : orig;
    half_t* dst = xside ? Xh : Yh;
    float*  nrm = xside ? X2 : Y2;
    half_t* o = dst + (size_t)p * KP;

    if (p >= NPATCH) {              // X-side padding rows: all zeros
        for (int f = 0; f < KP; ++f) o[f] = (half_t)0.f;
        nrm[p] = 0.f;
        return;
    }

    int pi = p / OW, pj = p % OW;
    float s = 0.f;
    int f = 0;
    for (int c = 0; c < 3; ++c)
        for (int di = 0; di < PATCH; ++di)
            for (int dj = 0; dj < PATCH; ++dj) {
                float v = src[(c * HW + pi + di) * HW + pj + dj];
                half_t h = (half_t)v;
                o[f++] = h;
                float hv = (float)h;
                s += hv * hv;
            }
    for (; f < KP; ++f) o[f] = (half_t)0.f;
    nrm[p] = s;
}

// ---------------------------------------------------------------------------
// Kernel 2: fused X@Y^T (WMMA f16->f32) + distance + row min/argmin.
// Block = 32-row strip of X (two 16-row A tiles sharing every B tile ->
// 6 WMMAs per B load). 8 waves sweep interleaved 1/8ths of the 961 column
// tiles. Min tracked as m2 = dist - x2 (row-constant shift) so the inner
// epilogue is fma + cmp + 2 cndmask per element. The 946 MB distance matrix
// is never materialized.
// ---------------------------------------------------------------------------
__global__ void __launch_bounds__(256)
nn_wmma_kernel(const half_t* __restrict__ Xh, const half_t* __restrict__ Yh,
               const float* __restrict__ X2, const float* __restrict__ Y2,
               float* __restrict__ nnd, float* __restrict__ out_idx) {
    const int wave = threadIdx.x >> 5;
    const int lane = threadIdx.x & 31;
    const int hsel = lane >> 4;   // which 16-lane half
    const int l16  = lane & 15;
    const int rowbase = blockIdx.x * 32;

    // --- load A (two 16x96 strips of X) once: contiguous k order, same as B -------
    // lanes 0-15 : row M=l16, K = {k0..k0+15}
    // lanes 16-31: row M=l16, K = {k0+16..k0+31}
    v16h a[2][3];
#pragma unroll
    for (int t = 0; t < 2; ++t) {
        const half_t* rp = Xh + (size_t)(rowbase + t * 16 + l16) * KP;
#pragma unroll
        for (int kc = 0; kc < 3; ++kc) {
            const half_t* q = rp + kc * 32 + hsel * 16;
            v8h lo = *(const v8h*)(q);
            v8h hi = *(const v8h*)(q + 8);
            v16h tt;
#pragma unroll
            for (int i = 0; i < 8; ++i) { tt[i] = lo[i]; tt[i + 8] = hi[i]; }
            a[t][kc] = tt;
        }
    }

    float m2[2][8];                 // running (min dist - x2[row])
    int   mini[2][8];
#pragma unroll
    for (int t = 0; t < 2; ++t)
#pragma unroll
        for (int r = 0; r < 8; ++r) { m2[t][r] = 3.0e38f; mini[t][r] = 0x7fffffff; }

    for (int j = wave; j < NT; j += 8) {
        const int colbase = j * 16;
        // B layout (32x16 f16): lane = column, lanes 0-15 hold K 0..15 of
        // the chunk, lanes 16-31 hold K 16..31 -> contiguous 32B per chunk
        const half_t* cp = Yh + (size_t)(colbase + l16) * KP + hsel * 16;
        v16h b0 = *(const v16h*)(cp);
        v16h b1 = *(const v16h*)(cp + 32);
        v16h b2 = *(const v16h*)(cp + 64);

        v8f cc[2] = {v8f{}, v8f{}};
        cc[0] = __builtin_amdgcn_wmma_f32_16x16x32_f16(false, a[0][0], false, b0, (short)0, cc[0], false, false);
        cc[1] = __builtin_amdgcn_wmma_f32_16x16x32_f16(false, a[1][0], false, b0, (short)0, cc[1], false, false);
        cc[0] = __builtin_amdgcn_wmma_f32_16x16x32_f16(false, a[0][1], false, b1, (short)0, cc[0], false, false);
        cc[1] = __builtin_amdgcn_wmma_f32_16x16x32_f16(false, a[1][1], false, b1, (short)0, cc[1], false, false);
        cc[0] = __builtin_amdgcn_wmma_f32_16x16x32_f16(false, a[0][2], false, b2, (short)0, cc[0], false, false);
        cc[1] = __builtin_amdgcn_wmma_f32_16x16x32_f16(false, a[1][2], false, b2, (short)0, cc[1], false, false);

        float y2  = Y2[colbase + l16];
        int   col = colbase + l16;
#pragma unroll
        for (int t = 0; t < 2; ++t)
#pragma unroll
            for (int r = 0; r < 8; ++r) {
                float d = __builtin_fmaf(-2.0f, cc[t][r], y2);  // dist - x2
                if (d < m2[t][r]) { m2[t][r] = d; mini[t][r] = col; }
            }
    }

    // --- reduce across the 16 lanes holding each row (xor masks <= 8 stay
    // inside each 16-lane half on wave32); tie-break on smaller index.
    // m2 comparisons == dist comparisons (same per-row constant shift). ----
#pragma unroll
    for (int m = 8; m >= 1; m >>= 1) {
#pragma unroll
        for (int t = 0; t < 2; ++t)
#pragma unroll
            for (int r = 0; r < 8; ++r) {
                float ov = __shfl_xor(m2[t][r], m, 32);
                int   oi = __shfl_xor(mini[t][r], m, 32);
                if (ov < m2[t][r] || (ov == m2[t][r] && oi < mini[t][r])) {
                    m2[t][r] = ov; mini[t][r] = oi;
                }
            }
    }

    __shared__ float smin[8][32];
    __shared__ int   sidx[8][32];
    if (l16 == 0) {
#pragma unroll
        for (int t = 0; t < 2; ++t)
#pragma unroll
            for (int r = 0; r < 8; ++r) {
                int rl = t * 16 + r + 8 * hsel;
                smin[wave][rl] = m2[t][r];
                sidx[wave][rl] = mini[t][r];
            }
    }
    __syncthreads();

    if (threadIdx.x < 32) {
        float bv = smin[0][threadIdx.x];
        int   bi = sidx[0][threadIdx.x];
#pragma unroll
        for (int w = 1; w < 8; ++w) {
            float v = smin[w][threadIdx.x];
            int   i = sidx[w][threadIdx.x];
            if (v < bv || (v == bv && i < bi)) { bv = v; bi = i; }
        }
        int row = rowbase + threadIdx.x;
        if (row < NPATCH) {
            nnd[row]     = bv + X2[row];   // undo the -x2 shift
            out_idx[row] = (float)bi;
        }
    }
}

// ---------------------------------------------------------------------------
// Kernel 3: mean of nn distances -> loss scalar.
// ---------------------------------------------------------------------------
__global__ void loss_kernel(const float* __restrict__ nnd,
                            float* __restrict__ out) {
    __shared__ float s[256];
    float acc = 0.f;
    for (int i = threadIdx.x; i < NPATCH; i += 256) acc += nnd[i];
    s[threadIdx.x] = acc;
    __syncthreads();
    for (int o = 128; o > 0; o >>= 1) {
        if ((int)threadIdx.x < o) s[threadIdx.x] += s[threadIdx.x + o];
        __syncthreads();
    }
    if (threadIdx.x == 0) out[0] = s[0] / (float)NPATCH;
}

// ---------------------------------------------------------------------------
static inline size_t align256(size_t x) { return (x + 255) & ~(size_t)255; }

extern "C" void kernel_launch(void* const* d_in, const int* in_sizes, int n_in,
                              void* d_out, int out_size, void* d_ws, size_t ws_size,
                              hipStream_t stream) {
    const float* crop = (const float*)d_in[0];
    const float* orig = (const float*)d_in[1];
    float* out = (float*)d_out;   // out[0] = loss, out[1..NPATCH] = nn indices

    // workspace partition (all offsets 256B aligned for 32B vector loads)
    char* ws = (char*)d_ws;
    size_t off = 0;
    half_t* Xh = (half_t*)(ws + off); off = align256(off + (size_t)NPAD   * KP * sizeof(half_t));
    half_t* Yh = (half_t*)(ws + off); off = align256(off + (size_t)NPATCH * KP * sizeof(half_t));
    float*  X2 = (float*)(ws + off);  off = align256(off + (size_t)NPAD   * sizeof(float));
    float*  Y2 = (float*)(ws + off);  off = align256(off + (size_t)NPATCH * sizeof(float));
    float*  nnd = (float*)(ws + off); off = align256(off + (size_t)NPATCH * sizeof(float));

    dim3 gU((NPAD + 255) / 256, 2);
    unfold_kernel<<<gU, 256, 0, stream>>>(crop, orig, Xh, Yh, X2, Y2);

    nn_wmma_kernel<<<dim3(NBLK), 256, 0, stream>>>(Xh, Yh, X2, Y2, nnd, out + 1);

    loss_kernel<<<dim3(1), 256, 0, stream>>>(nnd, out);
}
